// LinearRecurrentUnit_41034117546197
// MI455X (gfx1250) — compile-verified, matrix operand order checked by reading, not run on hardware
//
#include <hip/hip_runtime.h>
#include <hip/hip_bf16.h>

// ---------------------------------------------------------------------------
// LRU forward for MI455X (gfx1250, wave32, WMMA + async-to-LDS).
//   u    = x @ Wi^T + bi                     (32768 x 2048, K=1024)
//   g    = sigmoid(x @ Wg^T + bg)
//   h_t  = sigmoid(log_a) * h_{t-1} + u_t    (chunked parallel scan over T)
//   y    = (h * g) @ Wo^T + bo               (32768 x 1024, K=2048)
// GEMMs: v_wmma_f32_16x16x32_bf16, weight tiles staged to LDS with
// global_load_async_to_lds_b128 (ASYNCcnt) and double buffering.
// ---------------------------------------------------------------------------

#define B_   8
#define T_   4096
#define DIN_ 1024
#define H_   2048
#define BT_  (B_ * T_)
#define CH_  16              // scan chunks per sequence
#define CL_  (T_ / CH_)      // chunk length = 256 = 2^8

typedef __bf16 bf16_t;
typedef __attribute__((ext_vector_type(16))) __bf16 v16bf;
typedef __attribute__((ext_vector_type(8)))  __bf16 v8bf;
typedef __attribute__((ext_vector_type(8)))  float  v8f;

__device__ __forceinline__ float sigmoidf_(float x) {
    return 1.0f / (1.0f + __expf(-x));
}

// Fire-and-forget 16B global->LDS copy (ASYNCcnt-tracked).
// Generic LDS addresses keep the byte offset in the low 32 bits (ISA 10.2).
__device__ __forceinline__ void async_g2l_b128(const void* lds_ptr, const void* gptr) {
    unsigned lds_addr = (unsigned)(unsigned long long)lds_ptr;
    asm volatile("global_load_async_to_lds_b128 %0, %1, off"
                 :: "v"(lds_addr), "v"(gptr) : "memory");
}
__device__ __forceinline__ void wait_async0() {
    asm volatile("s_wait_asynccnt 0x0" ::: "memory");
}

// Load one 16x32 bf16 WMMA fragment from a row-major source (ld in elements).
// Per ISA 7.12.2: lanes 0-15 hold K {k0..k0+7, k0+16..k0+23} of row l16,
// lanes 16-31 hold K {k0+8..k0+15, k0+24..k0+31}.
__device__ __forceinline__ v16bf load_frag16x32(const bf16_t* __restrict__ base,
                                                int ld, int row, int k0, int half) {
    union { v16bf v; v8bf h[2]; } u;
    const bf16_t* p = base + (size_t)row * ld + k0 + half * 8;
    u.h[0] = *reinterpret_cast<const v8bf*>(p);
    u.h[1] = *reinterpret_cast<const v8bf*>(p + 16);
    return u.v;
}

// Same fragment pattern out of a staged LDS tile (64 rows x 32 K, row-major).
__device__ __forceinline__ v16bf lds_frag(const bf16_t* tile, int j, int l16, int half) {
    union { v16bf v; v8bf h[2]; } u;
    const bf16_t* p = tile + (j * 16 + l16) * 32 + half * 8;
    u.h[0] = *reinterpret_cast<const v8bf*>(p);        // ds_load_b128
    u.h[1] = *reinterpret_cast<const v8bf*>(p + 16);
    return u.v;
}

// ---------------------------------------------------------------------------
// Kernel 1: f32 -> bf16 convert (grid-stride)
// ---------------------------------------------------------------------------
__global__ void k_f32_to_bf16(const float* __restrict__ s, bf16_t* __restrict__ d, int n) {
    int i = blockIdx.x * blockDim.x + threadIdx.x;
    int stride = gridDim.x * blockDim.x;
    for (; i < n; i += stride) d[i] = (bf16_t)s[i];
}

// ---------------------------------------------------------------------------
// Kernel 2: fused u / gate GEMM with async double-buffered weight tiles.
// Block = 256 threads = 8 waves; wave w: rows [m0,m0+16), cols [n0,n0+64).
// Per k-step each thread issues 2 async b128 copies (Wi + Wg chunk).
// ---------------------------------------------------------------------------
__global__ void __launch_bounds__(256)
k_gemm_ug(const bf16_t* __restrict__ xb,  const bf16_t* __restrict__ wib,
          const bf16_t* __restrict__ wgb, const float* __restrict__ bi,
          const float* __restrict__ bg,   float* __restrict__ u,
          bf16_t* __restrict__ gate) {
    __shared__ __align__(16) bf16_t sWi[2][64 * 32];
    __shared__ __align__(16) bf16_t sWg[2][64 * 32];

    const int tid  = threadIdx.x;
    const int lane = tid & 31;
    const int wave = tid >> 5;
    const int l16  = lane & 15;
    const int half = lane >> 4;
    const int m0   = blockIdx.y * 128 + wave * 16;
    const int n0   = blockIdx.x * 64;

    // copy role: row = tid/4 (0..63), 16B chunk = tid%4
    const int crow = tid >> 2;
    const int cel  = (tid & 3) * 8;                 // element offset within 32-K row
    const bf16_t* gWi = wib + (size_t)(n0 + crow) * DIN_ + cel;
    const bf16_t* gWg = wgb + (size_t)(n0 + crow) * DIN_ + cel;
    bf16_t* lWi0 = &sWi[0][crow * 32 + cel];
    bf16_t* lWi1 = &sWi[1][crow * 32 + cel];
    bf16_t* lWg0 = &sWg[0][crow * 32 + cel];
    bf16_t* lWg1 = &sWg[1][crow * 32 + cel];

    v8f accU[4] = {};
    v8f accG[4] = {};

    // prologue: stage k0 = 0 into buffer 0
    async_g2l_b128(lWi0, gWi);
    async_g2l_b128(lWg0, gWg);

    for (int ks = 0; ks < DIN_ / 32; ++ks) {
        const int k0  = ks * 32;
        const int buf = ks & 1;
        wait_async0();          // my stage-ks copies have landed
        __syncthreads();        // everyone's copies landed; prev reads of buf^1 done
        if (k0 + 32 < DIN_) {   // stage next tile into the other buffer
            async_g2l_b128(buf ? lWi0 : lWi1, gWi + k0 + 32);
            async_g2l_b128(buf ? lWg0 : lWg1, gWg + k0 + 32);
        }
        v16bf a = load_frag16x32(xb, DIN_, m0 + l16, k0, half);
#pragma unroll
        for (int j = 0; j < 4; ++j) {
            v16bf bw = lds_frag(&sWi[buf][0], j, l16, half);
            accU[j] = __builtin_amdgcn_wmma_f32_16x16x32_bf16(
                false, a, false, bw, (short)0, accU[j], false, false);
            v16bf bgf = lds_frag(&sWg[buf][0], j, l16, half);
            accG[j] = __builtin_amdgcn_wmma_f32_16x16x32_bf16(
                false, a, false, bgf, (short)0, accG[j], false, false);
        }
    }

    // C/D layout: lane l, VGPR r -> n = n0 + (l&15), m = m0 + r + 8*(l>>4)
#pragma unroll
    for (int j = 0; j < 4; ++j) {
        const int n = n0 + j * 16 + l16;
        const float biv = bi[n];
        const float bgv = bg[n];
#pragma unroll
        for (int r = 0; r < 8; ++r) {
            const int m = m0 + r + 8 * half;
            const size_t idx = (size_t)m * H_ + n;
            u[idx]    = accU[j][r] + biv;
            gate[idx] = (bf16_t)sigmoidf_(accG[j][r] + bgv);
        }
    }
}

// ---------------------------------------------------------------------------
// Kernels 3a/3b/3c: chunked parallel scan.
//   p1: per (chunk,b,h) local scan from 0, store chunk-end state.
//   p2: per (b,h) sequential carry combine across 16 chunks (a^256 decay),
//       overwrite buffer with each chunk's INCOMING carry.
//   p3: per (chunk,b,h) rescan with incoming carry, write h*gate over gate.
// idx = (c*B + b)*H + h keeps h fastest -> coalesced.
// ---------------------------------------------------------------------------
__global__ void __launch_bounds__(256)
k_scan_p1(const float* __restrict__ u, const float* __restrict__ log_a,
          float* __restrict__ cend) {
    const int idx = blockIdx.x * blockDim.x + threadIdx.x;   // B*H*CH threads
    const int h  = idx & (H_ - 1);
    const int cb = idx >> 11;
    const int b  = cb & (B_ - 1);
    const int c  = cb >> 3;
    const float a = sigmoidf_(log_a[h]);
    size_t p = ((size_t)b * T_ + (size_t)c * CL_) * H_ + h;
    float hs = 0.0f;
    for (int t = 0; t < CL_; ++t) { hs = fmaf(a, hs, u[p]); p += H_; }
    cend[idx] = hs;
}

__global__ void __launch_bounds__(256)
k_scan_p2(const float* __restrict__ log_a, float* __restrict__ cend) {
    const int idx = blockIdx.x * blockDim.x + threadIdx.x;   // B*H threads
    const int h = idx & (H_ - 1);
    const int b = idx >> 11;
    const float a = sigmoidf_(log_a[h]);
    float aL = a;                       // a^(2^8) = a^CL_
#pragma unroll
    for (int i = 0; i < 8; ++i) aL *= aL;
    float carry = 0.0f;
#pragma unroll
    for (int c = 0; c < CH_; ++c) {
        const size_t q = ((size_t)c * B_ + b) * H_ + h;
        const float e = cend[q];
        cend[q] = carry;                // incoming state for chunk c
        carry = fmaf(aL, carry, e);
    }
}

__global__ void __launch_bounds__(256)
k_scan_p3(const float* __restrict__ u, const float* __restrict__ log_a,
          const float* __restrict__ cend, bf16_t* __restrict__ gh) {
    const int idx = blockIdx.x * blockDim.x + threadIdx.x;   // B*H*CH threads
    const int h  = idx & (H_ - 1);
    const int cb = idx >> 11;
    const int b  = cb & (B_ - 1);
    const int c  = cb >> 3;
    const float a = sigmoidf_(log_a[h]);
    size_t p = ((size_t)b * T_ + (size_t)c * CL_) * H_ + h;
    float hs = cend[idx];
    for (int t = 0; t < CL_; ++t) {
        hs = fmaf(a, hs, u[p]);
        const float g = (float)gh[p];
        gh[p] = (bf16_t)(hs * g);
        p += H_;
    }
}

// ---------------------------------------------------------------------------
// Kernel 4: output GEMM  y[m,d] = sum_h hg[m,h] * Wo[d,h] + bo[d]
// Same async double-buffered structure; 1 async copy/thread/stage, K = 2048.
// ---------------------------------------------------------------------------
__global__ void __launch_bounds__(256)
k_gemm_out(const bf16_t* __restrict__ hg, const bf16_t* __restrict__ wob,
           const float* __restrict__ bo, float* __restrict__ y) {
    __shared__ __align__(16) bf16_t sWo[2][64 * 32];

    const int tid  = threadIdx.x;
    const int lane = tid & 31;
    const int wave = tid >> 5;
    const int l16  = lane & 15;
    const int half = lane >> 4;
    const int m0   = blockIdx.y * 128 + wave * 16;
    const int n0   = blockIdx.x * 64;

    const int crow = tid >> 2;
    const int cel  = (tid & 3) * 8;
    const bf16_t* gWo = wob + (size_t)(n0 + crow) * H_ + cel;
    bf16_t* lWo0 = &sWo[0][crow * 32 + cel];
    bf16_t* lWo1 = &sWo[1][crow * 32 + cel];

    v8f acc[4] = {};

    async_g2l_b128(lWo0, gWo);
    for (int ks = 0; ks < H_ / 32; ++ks) {
        const int k0  = ks * 32;
        const int buf = ks & 1;
        wait_async0();
        __syncthreads();
        if (k0 + 32 < H_)
            async_g2l_b128(buf ? lWo0 : lWo1, gWo + k0 + 32);
        v16bf a = load_frag16x32(hg, H_, m0 + l16, k0, half);
#pragma unroll
        for (int j = 0; j < 4; ++j) {
            v16bf bw = lds_frag(&sWo[buf][0], j, l16, half);
            acc[j] = __builtin_amdgcn_wmma_f32_16x16x32_bf16(
                false, a, false, bw, (short)0, acc[j], false, false);
        }
    }

#pragma unroll
    for (int j = 0; j < 4; ++j) {
        const int n = n0 + j * 16 + l16;
        const float bov = bo[n];
#pragma unroll
        for (int r = 0; r < 8; ++r) {
            const int m = m0 + r + 8 * half;
            y[(size_t)m * DIN_ + n] = acc[j][r] + bov;
        }
    }
}

// ---------------------------------------------------------------------------
// Host-side launch
// ---------------------------------------------------------------------------
extern "C" void kernel_launch(void* const* d_in, const int* in_sizes, int n_in,
                              void* d_out, int out_size, void* d_ws, size_t ws_size,
                              hipStream_t stream) {
    const float* x     = (const float*)d_in[0];
    const float* Wi    = (const float*)d_in[1];
    const float* bi    = (const float*)d_in[2];
    const float* Wg    = (const float*)d_in[3];
    const float* bg    = (const float*)d_in[4];
    const float* Wo    = (const float*)d_in[5];
    const float* bo    = (const float*)d_in[6];
    const float* log_a = (const float*)d_in[7];
    float* y = (float*)d_out;

    // Workspace layout
    char* w = (char*)d_ws;
    size_t off = 0;
    bf16_t* xb   = (bf16_t*)(w + off); off += (size_t)BT_ * DIN_ * 2;   // 64 MiB
    bf16_t* wib  = (bf16_t*)(w + off); off += (size_t)H_ * DIN_ * 2;    //  4 MiB
    bf16_t* wgb  = (bf16_t*)(w + off); off += (size_t)H_ * DIN_ * 2;    //  4 MiB
    bf16_t* wob  = (bf16_t*)(w + off); off += (size_t)DIN_ * H_ * 2;    //  4 MiB
    float*  u    = (float*) (w + off); off += (size_t)BT_ * H_ * 4;     // 256 MiB
    bf16_t* gh   = (bf16_t*)(w + off); off += (size_t)BT_ * H_ * 2;     // 128 MiB
    float*  cend = (float*) (w + off); off += (size_t)CH_ * B_ * H_ * 4;//   1 MiB

    // 1) convert inputs to bf16
    k_f32_to_bf16<<<2048, 256, 0, stream>>>(x,  xb,  BT_ * DIN_);
    k_f32_to_bf16<<<512,  256, 0, stream>>>(Wi, wib, H_ * DIN_);
    k_f32_to_bf16<<<512,  256, 0, stream>>>(Wg, wgb, H_ * DIN_);
    k_f32_to_bf16<<<512,  256, 0, stream>>>(Wo, wob, DIN_ * H_);

    // 2) u / gate GEMM: grid (H/64, BT/128)
    k_gemm_ug<<<dim3(H_ / 64, BT_ / 128), 256, 0, stream>>>(xb, wib, wgb, bi, bg, u, gh);

    // 3) chunked scan
    k_scan_p1<<<(B_ * H_ * CH_) / 256, 256, 0, stream>>>(u, log_a, cend);
    k_scan_p2<<<(B_ * H_) / 256,       256, 0, stream>>>(log_a, cend);
    k_scan_p3<<<(B_ * H_ * CH_) / 256, 256, 0, stream>>>(u, log_a, cend, gh);

    // 4) output GEMM: grid (DIN/64, BT/128)
    k_gemm_out<<<dim3(DIN_ / 64, BT_ / 128), 256, 0, stream>>>(gh, wob, bo, y);
}
